// BatchHardTripletLoss_49855980372364
// MI455X (gfx1250) — matrix operand order repaired and checked by reading, not compiled, not run on hardware
//
#include <hip/hip_runtime.h>
#include <hip/hip_bf16.h>

// ---------------------------------------------------------------------------
// Batch-hard triplet loss, fused: never materializes the 8192x8192 distance
// matrix.
//   * Gram tiles via v_wmma_f32_16x16x32_bf16 (f32 accumulate), bf16 inputs.
//   * Hot loop reduces SQUARED distances (sqrt is monotonic -> one sqrt per
//     row in the O(N) epilogue instead of per N^2 element).
//   * has_pos / has_neg come from a 512-bin label histogram (O(N)), so the
//     N^2 loop carries no count accumulation at all.
//   * Each wave register-blocks a 32x16 output tile: one B fragment feeds
//     2 WMMAs.  __launch_bounds__(256) keeps A fragments register-resident.
// ---------------------------------------------------------------------------

typedef __attribute__((ext_vector_type(16))) __bf16 v16bf;
typedef __attribute__((ext_vector_type(8)))  float  v8f;

#define TD      128        // embedding dim (fixed by the reference)
#define TMARGIN 0.3f
#define JCHUNKS 8          // column-loop split for occupancy
#define RSTRIPS 2          // 16-row strips per wave (32 rows per wave)
#define NBINS   512        // label range per the reference

union ABFrag { v16bf v; uint4 q[2]; };

__device__ __forceinline__ unsigned short f2bf(float f) {
  unsigned int u = __float_as_uint(f);
  u += 0x7FFFu + ((u >> 16) & 1u);           // round to nearest even
  return (unsigned short)(u >> 16);
}

// One wave per row: fp32 -> bf16 copy + squared norm.
__global__ void __launch_bounds__(256)
prep_kernel(const float* __restrict__ x,
            unsigned short* __restrict__ bemb,
            float* __restrict__ sq, int N) {
  const int wave = (blockIdx.x * blockDim.x + threadIdx.x) >> 5;
  const int lane = threadIdx.x & 31;
  if (wave >= N) return;
  const float4 f = *(const float4*)(x + (size_t)wave * TD + lane * 4);
  unsigned int lo = (unsigned int)f2bf(f.x) | ((unsigned int)f2bf(f.y) << 16);
  unsigned int hi = (unsigned int)f2bf(f.z) | ((unsigned int)f2bf(f.w) << 16);
  *(uint2*)(bemb + (size_t)wave * TD + lane * 4) = make_uint2(lo, hi);
  float s = f.x * f.x + f.y * f.y + f.z * f.z + f.w * f.w;
#pragma unroll
  for (int m = 16; m >= 1; m >>= 1) s += __shfl_xor(s, m, 32);
  if (lane == 0) sq[wave] = s;
}

// Zero + build the label histogram (integer atomics -> deterministic).
__global__ void __launch_bounds__(256)
hist_zero_kernel(unsigned int* __restrict__ hist) {
  const int i = blockIdx.x * blockDim.x + threadIdx.x;
  if (i < NBINS) hist[i] = 0u;
}

__global__ void __launch_bounds__(256)
hist_kernel(const int* __restrict__ labels, unsigned int* __restrict__ hist,
            int N) {
  const int i = blockIdx.x * blockDim.x + threadIdx.x;
  if (i >= N) return;
  const unsigned b = (unsigned)labels[i];
  if (b < NBINS) atomicAdd(&hist[b], 1u);
}

// Each wave owns RSTRIPS 16-row strips and one column chunk.
// Partials are SQUARED distances.
__global__ void __launch_bounds__(256)
triplet_kernel(const unsigned short* __restrict__ bemb,
               const float* __restrict__ sq,
               const int* __restrict__ labels,
               float* __restrict__ pmax,   // [JCHUNKS][N] d^2 of hardest pos
               float* __restrict__ pmin,   // [JCHUNKS][N] d^2 of hardest neg
               int N) {
  const int lane  = threadIdx.x & 31;
  const int wave  = threadIdx.x >> 5;
  const int sgrp  = blockIdx.x * (blockDim.x >> 5) + wave;  // 32-row group id
  const int chunk = blockIdx.y;
  const int rowBase = sgrp * (16 * RSTRIPS);
  if (rowBase >= N) return;

  const int h  = lane >> 4;     // wave half (K sub-range / M half)
  const int lr = lane & 15;     // row (A) / column (B,C) within a tile

  // A fragments: RSTRIPS strips x 4 K-steps, resident in VGPRs.
  ABFrag aFrag[RSTRIPS][4];
  float  sqi[RSTRIPS][8];
  int    labi[RSTRIPS][8];
#pragma unroll
  for (int r = 0; r < RSTRIPS; ++r) {
    const int iRow = rowBase + r * 16 + lr;
#pragma unroll
    for (int ks = 0; ks < 4; ++ks) {
      aFrag[r][ks].q[0] = *(const uint4*)(bemb + (size_t)iRow * TD + ks * 32 + h * 8);
      aFrag[r][ks].q[1] = *(const uint4*)(bemb + (size_t)iRow * TD + ks * 32 + 16 + h * 8);
    }
#pragma unroll
    for (int v = 0; v < 8; ++v) {
      const int m = rowBase + r * 16 + 8 * h + v;   // C/D layout: m = v + 8*half
      sqi[r][v]  = sq[m];
      labi[r][v] = labels[m];
    }
  }

  float maxpos2[RSTRIPS][8], minneg2[RSTRIPS][8];
#pragma unroll
  for (int r = 0; r < RSTRIPS; ++r)
#pragma unroll
    for (int v = 0; v < 8; ++v) { maxpos2[r][v] = 0.f; minneg2[r][v] = 3.0e38f; }

  const int ntiles   = N >> 4;
  const int tilesPer = ntiles / JCHUNKS;
  const int jt0 = chunk * tilesPer;
  const int jt1 = jt0 + tilesPer;

  for (int jt = jt0; jt < jt1; ++jt) {
    const int   jRow = jt * 16 + lr;          // B column n = lr -> row jRow of x
    const float sqj  = sq[jRow];
    const int   labj = labels[jRow];
    v8f acc[RSTRIPS];
#pragma unroll
    for (int r = 0; r < RSTRIPS; ++r) acc[r] = (v8f){};
#pragma unroll
    for (int ks = 0; ks < 4; ++ks) {
      ABFrag b;                               // B: k = ks*32 + h*8 .. +7, then ks*32 + 16 + h*8 .. +7
      b.q[0] = *(const uint4*)(bemb + (size_t)jRow * TD + ks * 32 + h * 8);
      b.q[1] = *(const uint4*)(bemb + (size_t)jRow * TD + ks * 32 + 16 + h * 8);
#pragma unroll
      for (int r = 0; r < RSTRIPS; ++r)
        acc[r] = __builtin_amdgcn_wmma_f32_16x16x32_bf16(false, aFrag[r][ks].v,
                                                         false, b.v,
                                                         (short)0, acc[r],
                                                         false, false);
    }
    // Reduce on squared distances: sqrt hoisted out of the N^2 loop,
    // counts hoisted out entirely (label histogram).
#pragma unroll
    for (int r = 0; r < RSTRIPS; ++r)
#pragma unroll
      for (int v = 0; v < 8; ++v) {
        const float d2   = sqi[r][v] + sqj - 2.0f * acc[r][v];
        const bool  same = (labi[r][v] == labj);
        maxpos2[r][v] = fmaxf(maxpos2[r][v], same ? d2 : 0.f);  // == max(d2*same)
        minneg2[r][v] = fminf(minneg2[r][v], same ? 3.0e38f : d2);
      }
  }

  // Reduce across the 16 lanes of each half (rows live on one half).
#pragma unroll
  for (int r = 0; r < RSTRIPS; ++r)
#pragma unroll
    for (int v = 0; v < 8; ++v) {
#pragma unroll
      for (int m = 1; m <= 8; m <<= 1) {
        maxpos2[r][v] = fmaxf(maxpos2[r][v], __shfl_xor(maxpos2[r][v], m, 32));
        minneg2[r][v] = fminf(minneg2[r][v], __shfl_xor(minneg2[r][v], m, 32));
      }
    }
  if (lr == 0) {
#pragma unroll
    for (int r = 0; r < RSTRIPS; ++r)
#pragma unroll
      for (int v = 0; v < 8; ++v) {
        const int m = rowBase + r * 16 + 8 * h + v;
        pmax[(size_t)chunk * N + m] = maxpos2[r][v];
        pmin[(size_t)chunk * N + m] = minneg2[r][v];
      }
  }
}

// Combine chunk partials per row; sqrt applied ONCE per row here (O(N)).
// Validity from the label histogram: has_pos = mult>=2, has_neg = mult<N.
__global__ void __launch_bounds__(256)
combine_kernel(const float* __restrict__ pmax,
               const float* __restrict__ pmin,
               const int* __restrict__ labels,
               const unsigned int* __restrict__ hist,
               float* __restrict__ lossArr,
               float* __restrict__ validArr, int N) {
  const int i = blockIdx.x * blockDim.x + threadIdx.x;
  if (i >= N) return;
  float mp = 0.f, mn = 3.0e38f;
#pragma unroll
  for (int c = 0; c < JCHUNKS; ++c) {
    mp = fmaxf(mp, pmax[(size_t)c * N + i]);
    mn = fminf(mn, pmin[(size_t)c * N + i]);
  }
  const unsigned b   = (unsigned)labels[i];
  const unsigned mult = (b < NBINS) ? hist[b] : 0u;
  const bool valid = (mult >= 2u) && (mult < (unsigned)N);  // has_pos && has_neg
  const float hp = sqrtf(fmaxf(mp, 0.f));                   // dist = sqrt(max(d2,0))
  const float hn = sqrtf(fmaxf(mn, 0.f));
  float l = hp - hn + TMARGIN;
  l = l > 0.f ? l : 0.f;
  lossArr[i]  = valid ? l : 0.f;
  validArr[i] = valid ? 1.f : 0.f;
}

// Single-block deterministic reduction -> scalar mean over valid rows.
__global__ void __launch_bounds__(256)
finalize_kernel(const float* __restrict__ lossArr,
                const float* __restrict__ validArr,
                float* __restrict__ out, int N) {
  __shared__ float sl[256];
  __shared__ float sv[256];
  float a = 0.f, b = 0.f;
  for (int i = threadIdx.x; i < N; i += 256) { a += lossArr[i]; b += validArr[i]; }
  sl[threadIdx.x] = a; sv[threadIdx.x] = b;
  __syncthreads();
  for (int s = 128; s > 0; s >>= 1) {
    if (threadIdx.x < s) {
      sl[threadIdx.x] += sl[threadIdx.x + s];
      sv[threadIdx.x] += sv[threadIdx.x + s];
    }
    __syncthreads();
  }
  if (threadIdx.x == 0) {
    const float cnt = sv[0], s = sl[0];
    out[0] = cnt > 0.f ? s / fmaxf(cnt, 1.f) : s / (float)N;
  }
}

extern "C" void kernel_launch(void* const* d_in, const int* in_sizes, int n_in,
                              void* d_out, int out_size, void* d_ws, size_t ws_size,
                              hipStream_t stream) {
  (void)n_in; (void)out_size; (void)ws_size;
  const float* emb    = (const float*)d_in[0];   // [N,128] fp32
  const int*   labels = (const int*)d_in[1];     // [N] int
  float*       out    = (float*)d_out;           // scalar
  const int N = in_sizes[1];                     // 8192

  // Workspace layout (all naturally aligned):
  char* ws = (char*)d_ws;
  unsigned short* bemb = (unsigned short*)ws;                 // N*128 bf16  (2 MB)
  size_t off = (size_t)N * TD * sizeof(unsigned short);
  float* sq        = (float*)(ws + off); off += (size_t)N * sizeof(float);
  float* pmax      = (float*)(ws + off); off += (size_t)JCHUNKS * N * sizeof(float);
  float* pmin      = (float*)(ws + off); off += (size_t)JCHUNKS * N * sizeof(float);
  unsigned int* hist = (unsigned int*)(ws + off); off += NBINS * sizeof(unsigned int);
  float* lossArr   = (float*)(ws + off); off += (size_t)N * sizeof(float);
  float* validArr  = (float*)(ws + off);

  // 1) bf16 copy + squared norms (wave per row) and label histogram.
  prep_kernel<<<(N + 7) / 8, 256, 0, stream>>>(emb, bemb, sq, N);
  hist_zero_kernel<<<(NBINS + 255) / 256, 256, 0, stream>>>(hist);
  hist_kernel<<<(N + 255) / 256, 256, 0, stream>>>(labels, hist, N);

  // 2) fused WMMA distance + batch-hard reductions on squared distances.
  const int sgroups = N / (16 * RSTRIPS);         // 256
  dim3 grid((sgroups + 7) / 8, JCHUNKS);          // 32 x 8 blocks, 8 waves each
  triplet_kernel<<<grid, 256, 0, stream>>>(bemb, sq, labels, pmax, pmin, N);

  // 3) combine chunk partials -> per-row loss (sqrt + histogram validity).
  combine_kernel<<<(N + 255) / 256, 256, 0, stream>>>(pmax, pmin, labels, hist,
                                                      lossArr, validArr, N);

  // 4) deterministic scalar reduction.
  finalize_kernel<<<1, 256, 0, stream>>>(lossArr, validArr, out, N);
}
